// CrissCrossAttention_76914274336927
// MI455X (gfx1250) — matrix operand order repaired and checked by reading, compiled, not run on hardware
//
#include <hip/hip_runtime.h>
#include <hip/hip_bf16.h>

typedef __attribute__((ext_vector_type(16))) __bf16 v16bf;
typedef __attribute__((ext_vector_type(8)))  __bf16 v8bf;
typedef __attribute__((ext_vector_type(8)))  float  v8f;

#define BB 8
#define CC 512
#define CQ 64
#define HH 96
#define WW 96
#define HWP 9216            // H*W
#define NP  73728           // B*H*W
#define NS  768             // B*96 slices

__device__ inline v16bf mk16(v8bf lo, v8bf hi) {
  return __builtin_shufflevector(lo, hi, 0,1,2,3,4,5,6,7,8,9,10,11,12,13,14,15);
}
__device__ inline v8f wmma_bf16(v16bf a, v16bf b, v8f c) {
  return __builtin_amdgcn_wmma_f32_16x16x32_bf16(false, a, false, b, (short)0, c, false, false);
}
__device__ inline v8f zero8() { v8f z = {0.f,0.f,0.f,0.f,0.f,0.f,0.f,0.f}; return z; }

// ---------------------------------------------------------------------------
// Tiled transpose + f32->bf16 convert.  in: [Z][R][Cc] -> out: [Z][Cc][R]
// ---------------------------------------------------------------------------
__global__ void __launch_bounds__(256)
transpose_f32_bf16(const float* __restrict__ in, __bf16* __restrict__ out,
                   int R, int Cc) {
  __shared__ float tile[32][33];
  const int z  = blockIdx.z;
  const float* inb = in  + (size_t)z * R * Cc;
  __bf16*     outb = out + (size_t)z * R * Cc;
  const int c0 = blockIdx.x * 32, r0 = blockIdx.y * 32;
  const int tx = threadIdx.x, ty = threadIdx.y;
#pragma unroll
  for (int i = 0; i < 4; i++)
    tile[ty + i * 8][tx] = inb[(size_t)(r0 + ty + i * 8) * Cc + c0 + tx];
  __syncthreads();
#pragma unroll
  for (int i = 0; i < 4; i++)
    outb[(size_t)(c0 + ty + i * 8) * R + r0 + tx] = (__bf16)tile[tx][ty + i * 8];
}

// ---------------------------------------------------------------------------
// Elementwise f32 -> bf16 convert (weights).
// ---------------------------------------------------------------------------
__global__ void __launch_bounds__(256)
convert_f32_bf16(const float* __restrict__ in, __bf16* __restrict__ out, int n) {
  int i = blockIdx.x * 256 + threadIdx.x;
  if (i < n) out[i] = (__bf16)in[i];
}

// ---------------------------------------------------------------------------
// 1x1 conv GEMM, LDS-free:  D = xt[b] (9216x512) * Wb^T + bias
//   A-frag: xt pixel rows (contiguous K).  B-frag: Wb[o][c] rows (contiguous K).
// Wave computes 2 M-tiles sharing one B-frag. Block = 8 waves = 64M x 64N.
// Stores: dstH idx = (b*96+w)*(96*O) + h*hs + c*cs   (and w<->h for dstW)
//   q/k: hs=O, cs=1  (pixel-major [B,96,96,O]) ; v: hs=1, cs=96 ([B,96,O,96])
// ---------------------------------------------------------------------------
__global__ void __launch_bounds__(256)
conv1x1_wmma(const __bf16* __restrict__ xt,   // [B][9216][512]
             const __bf16* __restrict__ Wb,   // [O][512]
             const float*  __restrict__ bias, // [O]
             __bf16* __restrict__ dstH,
             __bf16* __restrict__ dstW,
             int O, int hs, int cs) {
  const int b    = blockIdx.z;
  const int mb   = blockIdx.x * 64;
  const int nbG  = blockIdx.y * 64;
  const int tid  = threadIdx.x;
  const int lane = tid & 31, wid = tid >> 5;
  const int mt = wid & 1, nt = wid >> 1;

  const int lrow   = lane & 15;
  const int runoff = (lane >> 4) * 8;
  const int koff   = (lane >> 4) * 8;
  const int m0     = mb + mt * 32;
  const int nglob  = nbG + nt * 16 + lrow;

  const __bf16* arow0 = xt + ((size_t)b * HWP + m0 + lrow) * CC;
  const __bf16* arow1 = arow0 + (size_t)16 * CC;
  const __bf16* brow  = Wb + (size_t)nglob * CC;

  v8f acc0 = zero8(), acc1 = zero8();
#pragma unroll 4
  for (int kb = 0; kb < CC; kb += 32) {
    __builtin_prefetch((const void*)(arow0 + kb + 64), 0, 1);
    __builtin_prefetch((const void*)(arow1 + kb + 64), 0, 1);
    v8bf b0 = *(const v8bf*)(brow + kb + koff);
    v8bf b1 = *(const v8bf*)(brow + kb + koff + 16);
    v16bf bf = mk16(b0, b1);
    v8bf a00 = *(const v8bf*)(arow0 + kb + runoff);
    v8bf a01 = *(const v8bf*)(arow0 + kb + 16 + runoff);
    acc0 = wmma_bf16(mk16(a00, a01), bf, acc0);
    v8bf a10 = *(const v8bf*)(arow1 + kb + runoff);
    v8bf a11 = *(const v8bf*)(arow1 + kb + 16 + runoff);
    acc1 = wmma_bf16(mk16(a10, a11), bf, acc1);
  }

  const float bs = bias[nglob];
  const int   rb = (lane >> 4) * 8;
  const size_t span = (size_t)96 * O;
  const int   c  = nglob;

  auto store_tile = [&](v8f acc, int pbase) {
    int h0 = pbase / WW;
    int w0 = pbase - h0 * WW;
#pragma unroll
    for (int r = 0; r < 8; r++) {
      int w = w0 + r, h = h0;
      if (w >= WW) { w -= WW; h += 1; }
      __bf16 v = (__bf16)(acc[r] + bs);
      dstH[(size_t)(b * 96 + w) * span + (size_t)h * hs + (size_t)c * cs] = v;
      dstW[(size_t)(b * 96 + h) * span + (size_t)w * hs + (size_t)c * cs] = v;
    }
  };
  store_tile(acc0, m0 + rb);
  store_tile(acc1, m0 + 16 + rb);
}

// ---------------------------------------------------------------------------
// Energy GEMM, LDS-free: per slice s, E[m][n] = sum_c Q[m][c] * K[n][c]
// q,k: [S][96][64] bf16 (pixel-major).  f32 energies scattered into
// att [B,H,W,192]:  off = (s/96)*1769472 + (s%96)*SI + m*MR + n + NO
// ---------------------------------------------------------------------------
__global__ void __launch_bounds__(256)
energy_wmma(const __bf16* __restrict__ q, const __bf16* __restrict__ k,
            float* __restrict__ att, int SI, int MR, int NO) {
  const int s   = blockIdx.x;
  const int tid = threadIdx.x;
  const int lane = tid & 31, wid = tid >> 5;
  const __bf16* qs = q + (size_t)s * (96 * CQ);
  const __bf16* ks = k + (size_t)s * (96 * CQ);
  const int lrow   = lane & 15;
  const int runoff = (lane >> 4) * 8;
  const int koff   = (lane >> 4) * 8;
  const size_t obase = (size_t)(s / 96) * 1769472 + (size_t)(s % 96) * SI;

  for (int t = wid; t < 36; t += 8) {
    const int mtb = (t / 6) * 16, ntb = (t % 6) * 16;
    const __bf16* ar = qs + (size_t)(mtb + lrow) * CQ;
    const __bf16* br = ks + (size_t)(ntb + lrow) * CQ;
    v8f acc = zero8();
#pragma unroll
    for (int kb = 0; kb < CQ; kb += 32) {
      v8bf a0 = *(const v8bf*)(ar + kb + runoff);
      v8bf a1 = *(const v8bf*)(ar + kb + 16 + runoff);
      v8bf b0 = *(const v8bf*)(br + kb + koff);
      v8bf b1 = *(const v8bf*)(br + kb + koff + 16);
      acc = wmma_bf16(mk16(a0, a1), mk16(b0, b1), acc);
    }
    const int rb = (lane >> 4) * 8;
#pragma unroll
    for (int r = 0; r < 8; r++)
      att[obase + (size_t)(mtb + r + rb) * MR + (ntb + lrow) + NO] = acc[r];
  }
}

// ---------------------------------------------------------------------------
// Softmax over 192 with diagonal mask on first 96 (j == h -> -inf).
// ---------------------------------------------------------------------------
__global__ void __launch_bounds__(256)
softmax_mask(const float* __restrict__ att, __bf16* __restrict__ abf) {
  const int row  = blockIdx.x * 8 + (threadIdx.x >> 5);
  const int lane = threadIdx.x & 31;
  const int h = (row / WW) % HH;
  const float* a = att + (size_t)row * 192;
  float v[6];
  float m = -3.0e38f;
#pragma unroll
  for (int i = 0; i < 6; i++) {
    int j = lane + i * 32;
    float x = a[j];
    if (j < 96 && j == h) x = -3.0e38f;
    v[i] = x;
    m = fmaxf(m, x);
  }
  for (int off = 16; off > 0; off >>= 1) m = fmaxf(m, __shfl_xor(m, off, 32));
  float sum = 0.f;
#pragma unroll
  for (int i = 0; i < 6; i++) { v[i] = __expf(v[i] - m); sum += v[i]; }
  for (int off = 16; off > 0; off >>= 1) sum += __shfl_xor(sum, off, 32);
  float inv = 1.0f / sum;
  __bf16* o = abf + (size_t)row * 192;
#pragma unroll
  for (int i = 0; i < 6; i++) o[lane + i * 32] = (__bf16)(v[i] * inv);
}

// ---------------------------------------------------------------------------
// Aggregation GEMM, LDS-free: per slice s, O[c][n] = sum_k V[c][k] * A[n][k]
// Wave computes 2 C-tiles sharing one attention B-frag (96 units / 12 blocks).
//   A row: abf + (s/96)*1769472 + (s%96)*ASI + n*AN + AO + k
//   out idx = (s/96)*4718592 + c*9216 + (s%96)*OSI + n*ON
// pass1 (resid!=null): out = g*acc + resid ; pass2: out += g*acc
// ---------------------------------------------------------------------------
__global__ void __launch_bounds__(256)
aggregate_wmma(const __bf16* __restrict__ vsrc,   // [S][512][96]
               const __bf16* __restrict__ abf,
               const float*  __restrict__ resid,
               float* __restrict__ out,
               const float* __restrict__ gammap,
               int ASI, int AN, int AO, int OSI, int ON) {
  const int s    = blockIdx.y;
  const int tid  = threadIdx.x;
  const int lane = tid & 31;
  const int u    = blockIdx.x * 8 + (tid >> 5);   // 0..95
  const int mp   = u / 6;                         // C tile pair (0..15)
  const int ntb  = (u % 6) * 16;

  const __bf16* vs = vsrc + (size_t)s * (CC * 96);
  const size_t abase = (size_t)(s / 96) * 1769472 + (size_t)(s % 96) * ASI + AO;
  const int lrow   = lane & 15;
  const int runoff = (lane >> 4) * 8;
  const int koff   = (lane >> 4) * 8;
  const __bf16* ar0 = vs + (size_t)(mp * 32 + lrow) * 96;
  const __bf16* ar1 = ar0 + (size_t)16 * 96;
  const __bf16* ap  = abf + abase + (size_t)(ntb + lrow) * AN;

  v8f acc0 = zero8(), acc1 = zero8();
#pragma unroll
  for (int kb = 0; kb < 96; kb += 32) {
    v8bf b0 = *(const v8bf*)(ap + kb + koff);
    v8bf b1 = *(const v8bf*)(ap + kb + koff + 16);
    v16bf bf = mk16(b0, b1);
    v8bf a00 = *(const v8bf*)(ar0 + kb + runoff);
    v8bf a01 = *(const v8bf*)(ar0 + kb + 16 + runoff);
    acc0 = wmma_bf16(mk16(a00, a01), bf, acc0);
    v8bf a10 = *(const v8bf*)(ar1 + kb + runoff);
    v8bf a11 = *(const v8bf*)(ar1 + kb + 16 + runoff);
    acc1 = wmma_bf16(mk16(a10, a11), bf, acc1);
  }

  const float g = gammap[0];
  const size_t ob = (size_t)(s / 96) * 4718592 + (size_t)(s % 96) * OSI;
  const int rb = (lane >> 4) * 8;
  const int n  = ntb + lrow;

  auto store_tile = [&](v8f acc, int cbase) {
#pragma unroll
    for (int r = 0; r < 8; r++) {
      size_t idx = ob + (size_t)(cbase + r + rb) * HWP + (size_t)n * ON;
      if (resid) out[idx] = g * acc[r] + resid[idx];
      else       out[idx] += g * acc[r];
    }
  };
  store_tile(acc0, mp * 32);
  store_tile(acc1, mp * 32 + 16);
}

// ---------------------------------------------------------------------------
extern "C" void kernel_launch(void* const* d_in, const int* in_sizes, int n_in,
                              void* d_out, int out_size, void* d_ws, size_t ws_size,
                              hipStream_t stream) {
  const float* x     = (const float*)d_in[0];
  const float* y     = (const float*)d_in[1];
  const float* Wq    = (const float*)d_in[2];
  const float* bq    = (const float*)d_in[3];
  const float* Wk    = (const float*)d_in[4];
  const float* bk    = (const float*)d_in[5];
  const float* Wv    = (const float*)d_in[6];
  const float* bv    = (const float*)d_in[7];
  const float* gamma = (const float*)d_in[8];
  float* out1 = (float*)d_out;
  float* out2 = out1 + (size_t)BB * CC * HH * WW;

  char* ws = (char*)d_ws;
  size_t off = 0;
  auto take = [&](size_t bytes) {
    size_t r = off;
    off += (bytes + 255) & ~(size_t)255;
    return r;
  };
  const size_t nBCHW = (size_t)BB * CC * HH * WW;
  const size_t nQ    = (size_t)NS * 96 * CQ;
  const size_t nAtt  = (size_t)NP * 192;

  __bf16* xt   = (__bf16*)(ws + take(nBCHW * 2));
  __bf16* yt   = (__bf16*)(ws + take(nBCHW * 2));
  __bf16* Wqb  = (__bf16*)(ws + take((size_t)CQ * CC * 2));
  __bf16* Wkb  = (__bf16*)(ws + take((size_t)CQ * CC * 2));
  __bf16* Wvb  = (__bf16*)(ws + take((size_t)CC * CC * 2));
  __bf16* qxH  = (__bf16*)(ws + take(nQ * 2));
  __bf16* qxW  = (__bf16*)(ws + take(nQ * 2));
  __bf16* qyH  = (__bf16*)(ws + take(nQ * 2));
  __bf16* qyW  = (__bf16*)(ws + take(nQ * 2));
  __bf16* kxH  = (__bf16*)(ws + take(nQ * 2));
  __bf16* kxW  = (__bf16*)(ws + take(nQ * 2));
  __bf16* kyH  = (__bf16*)(ws + take(nQ * 2));
  __bf16* kyW  = (__bf16*)(ws + take(nQ * 2));
  __bf16* vxH  = (__bf16*)(ws + take(nBCHW * 2));
  __bf16* vxW  = (__bf16*)(ws + take(nBCHW * 2));
  __bf16* vyH  = (__bf16*)(ws + take(nBCHW * 2));
  __bf16* vyW  = (__bf16*)(ws + take(nBCHW * 2));
  float*  att_xy = (float*)(ws + take(nAtt * 4));
  float*  att_yx = (float*)(ws + take(nAtt * 4));
  __bf16* abf_xy = (__bf16*)(ws + take(nAtt * 2));
  __bf16* abf_yx = (__bf16*)(ws + take(nAtt * 2));
  (void)ws_size; (void)in_sizes; (void)n_in; (void)out_size;

  dim3 tb(32, 8, 1);
  // x,y: per batch [512][9216] -> [9216][512] bf16
  transpose_f32_bf16<<<dim3(HWP / 32, CC / 32, BB), tb, 0, stream>>>(x, xt, CC, HWP);
  transpose_f32_bf16<<<dim3(HWP / 32, CC / 32, BB), tb, 0, stream>>>(y, yt, CC, HWP);
  // weights: straight bf16 copy, keep [O][C] (that IS the B-frag layout)
  convert_f32_bf16<<<(CQ * CC + 255) / 256, 256, 0, stream>>>(Wq, Wqb, CQ * CC);
  convert_f32_bf16<<<(CQ * CC + 255) / 256, 256, 0, stream>>>(Wk, Wkb, CQ * CC);
  convert_f32_bf16<<<(CC * CC + 255) / 256, 256, 0, stream>>>(Wv, Wvb, CC * CC);

  // 1x1 convs. q/k -> pixel-major [B,96,96,64] (hs=64, cs=1)
  //            v   -> channel-major [B,96,512,96] (hs=1, cs=96)
  conv1x1_wmma<<<dim3(HWP / 64, 1, BB), 256, 0, stream>>>(xt, Wqb, bq, qxH, qxW, CQ, CQ, 1);
  conv1x1_wmma<<<dim3(HWP / 64, 1, BB), 256, 0, stream>>>(yt, Wqb, bq, qyH, qyW, CQ, CQ, 1);
  conv1x1_wmma<<<dim3(HWP / 64, 1, BB), 256, 0, stream>>>(xt, Wkb, bk, kxH, kxW, CQ, CQ, 1);
  conv1x1_wmma<<<dim3(HWP / 64, 1, BB), 256, 0, stream>>>(yt, Wkb, bk, kyH, kyW, CQ, CQ, 1);
  conv1x1_wmma<<<dim3(HWP / 64, CC / 64, BB), 256, 0, stream>>>(xt, Wvb, bv, vxH, vxW, CC, 1, 96);
  conv1x1_wmma<<<dim3(HWP / 64, CC / 64, BB), 256, 0, stream>>>(yt, Wvb, bv, vyH, vyW, CC, 1, 96);

  // energies: eH (SI=192, MR=18432, NO=0); eW (SI=18432, MR=192, NO=96)
  energy_wmma<<<NS, 256, 0, stream>>>(qxH, kyH, att_xy, 192, 18432, 0);
  energy_wmma<<<NS, 256, 0, stream>>>(qxW, kyW, att_xy, 18432, 192, 96);
  energy_wmma<<<NS, 256, 0, stream>>>(qyH, kxH, att_yx, 192, 18432, 0);
  energy_wmma<<<NS, 256, 0, stream>>>(qyW, kxW, att_yx, 18432, 192, 96);

  // masked softmax -> bf16 attention
  softmax_mask<<<NP / 8, 256, 0, stream>>>(att_xy, abf_xy);
  softmax_mask<<<NP / 8, 256, 0, stream>>>(att_yx, abf_yx);

  // out1 = g*(vx·attH_yx) + x ; then += g*(vx·attW_yx)
  aggregate_wmma<<<dim3(12, NS), 256, 0, stream>>>(vxH, abf_yx, x, out1, gamma,
                                                   192, 18432, 0, 1, 96);
  aggregate_wmma<<<dim3(12, NS), 256, 0, stream>>>(vxW, abf_yx, nullptr, out1, gamma,
                                                   18432, 192, 96, 96, 1);
  // out2 = g*(vy·attH_xy) + y ; then += g*(vy·attW_xy)
  aggregate_wmma<<<dim3(12, NS), 256, 0, stream>>>(vyH, abf_xy, y, out2, gamma,
                                                   192, 18432, 0, 1, 96);
  aggregate_wmma<<<dim3(12, NS), 256, 0, stream>>>(vyW, abf_xy, nullptr, out2, gamma,
                                                   18432, 192, 96, 1, 1);
}
